// ContraGraphLearning_34677565948079
// MI455X (gfx1250) — matrix operand order repaired and checked by reading, not compiled, not run on hardware
//
#include <hip/hip_runtime.h>
#include <hip/hip_bf16.h>
#include <math.h>

typedef __bf16 bf16t;
typedef __attribute__((ext_vector_type(16))) __bf16 bf16x16;
typedef __attribute__((ext_vector_type(8)))  float  f32x8;

union pack4 { unsigned long long u; bf16t h[4]; };

// ---------------------------------------------------------------------------
// Generic batched GEMM: C[b] = alpha * A[b](MxK,bf16) * B[b](KxN,bf16)
//                              [+ C_prev] [+ beta*D] [+ bias] [relu]
// Preconditions (guaranteed by caller via padded buffers):
//   - K is a multiple of 32; rows 0..K-1 of B and cols 0..K-1 of A readable,
//     with zeros on at least one side of every K-pair in the padded range.
//   - A rows clamped to M-1, B col-chunks clamped to N-4 (N multiple of 4).
// Block tile 64x64, 4 waves each computing 32x32 (2x2 WMMA 16x16x32 tiles).
// Staging is fully branch-free vector b64 loads.
// ---------------------------------------------------------------------------
template<bool OUT_BF16, bool ACC, bool HAS_D, bool HAS_BIAS, bool RELU>
__global__ __launch_bounds__(128)
void gemm_bf16_wmma(int M, int N, int K,
                    const bf16t* __restrict__ Ag, long long sA, int lda,
                    const bf16t* __restrict__ Bg, long long sB, int ldb,
                    void*        __restrict__ Cg, long long sC, int ldc,
                    const bf16t* __restrict__ Dg, long long sD, int ldd,
                    const float* __restrict__ bias,
                    float alpha, float beta)
{
    __shared__ bf16t As[64][36];    // [m][k]   (36: rows stay 8B-aligned)
    __shared__ bf16t BsT[64][36];   // [n][k]   (B tile stored transposed)

    const int bz = blockIdx.z;
    const int n0 = blockIdx.x * 64;
    const int m0 = blockIdx.y * 64;
    const bf16t* Ab = Ag + (long long)bz * sA;
    const bf16t* Bb = Bg + (long long)bz * sB;

    const int tid  = threadIdx.x;
    const int lane = tid & 31;
    const int w    = tid >> 5;
    const int wm   = w >> 1, wn = w & 1;
    const int half = lane >> 4, l16 = lane & 15;
    const int Mm1  = M - 1;
    const int Nm4  = N - 4;

    f32x8 acc[2][2] = {};

    for (int ks = 0; ks < K; ks += 32) {
        // ---- stage A tile (64x32): rows clamped, K always in (padded) range ----
        #pragma unroll
        for (int t = 0; t < 4; ++t) {
            int idx = (tid + t * 128) * 4;
            int r = idx >> 5, c = idx & 31;
            int gm = m0 + r;
            gm = gm < Mm1 ? gm : Mm1;
            *(unsigned long long*)&As[r][c] =
                *(const unsigned long long*)(Ab + (long long)gm * lda + (ks + c));
        }
        // ---- stage B tile (32x64) transposed; col chunk base clamped ----
        #pragma unroll
        for (int t = 0; t < 4; ++t) {
            int idx = (tid + t * 128) * 4;
            int kk = idx >> 6, nn = idx & 63;
            int gn = n0 + nn;
            gn = gn < Nm4 ? gn : Nm4;
            pack4 p;
            p.u = *(const unsigned long long*)(Bb + (long long)(ks + kk) * ldb + gn);
            BsT[nn + 0][kk] = p.h[0];
            BsT[nn + 1][kk] = p.h[1];
            BsT[nn + 2][kk] = p.h[2];
            BsT[nn + 3][kk] = p.h[3];
        }
        __syncthreads();

        bf16x16 afrag[2], bfrag[2];
        // A fragment: lane L holds row M=L%16; VGPR v pairs: K = (v>=4?16:0)+(half?8:0)+2*(v%4)
        #pragma unroll
        for (int ti = 0; ti < 2; ++ti) {
            int am = wm * 32 + ti * 16 + l16;
            #pragma unroll
            for (int v = 0; v < 8; ++v) {
                int kp = ((v & 4) << 2) + (half << 3) + ((v & 3) << 1);
                afrag[ti][2 * v]     = As[am][kp];
                afrag[ti][2 * v + 1] = As[am][kp + 1];
            }
        }
        // B fragment: lane L holds col N=L%16; VGPR v pairs: K = (v>=4?16:0)+(half?8:0)+2*(v%4)
        #pragma unroll
        for (int tj = 0; tj < 2; ++tj) {
            int bn = wn * 32 + tj * 16 + l16;
            #pragma unroll
            for (int v = 0; v < 8; ++v) {
                int kp = ((v & 4) << 2) + (half << 3) + ((v & 3) << 1);
                bfrag[tj][2 * v]     = BsT[bn][kp];
                bfrag[tj][2 * v + 1] = BsT[bn][kp + 1];
            }
        }
        #pragma unroll
        for (int ti = 0; ti < 2; ++ti)
            #pragma unroll
            for (int tj = 0; tj < 2; ++tj)
                acc[ti][tj] = __builtin_amdgcn_wmma_f32_16x16x32_bf16(
                    false, afrag[ti], false, bfrag[tj], (short)0, acc[ti][tj], false, false);
        __syncthreads();
    }

    // ---- epilogue (C/D layout: VGPR j -> row j (lanes 0-15) / 8+j (lanes 16-31)) ----
    #pragma unroll
    for (int ti = 0; ti < 2; ++ti)
        #pragma unroll
        for (int tj = 0; tj < 2; ++tj)
            #pragma unroll
            for (int j = 0; j < 8; ++j) {
                int row = m0 + wm * 32 + ti * 16 + (half << 3) + j;
                int col = n0 + wn * 32 + tj * 16 + l16;
                if (row < M && col < N) {
                    long long off = (long long)bz * sC + (long long)row * ldc + col;
                    float v = alpha * acc[ti][tj][j];
                    if (ACC)      v += ((const float*)Cg)[off];
                    if (HAS_D)    v += beta * (float)Dg[(long long)bz * sD + (long long)row * ldd + col];
                    if (HAS_BIAS) v += bias[col];
                    if (RELU)     v = fmaxf(v, 0.0f);
                    if (OUT_BF16) ((bf16t*)Cg)[off] = (bf16t)v;
                    else          ((float*)Cg)[off] = v;
                }
            }
}

// ---------------------------------------------------------------------------
// Helpers
// ---------------------------------------------------------------------------
__global__ void zero_f32_kernel(float* __restrict__ p, long long n)
{
    long long stride = (long long)gridDim.x * blockDim.x;
    for (long long i = blockIdx.x * (long long)blockDim.x + threadIdx.x; i < n; i += stride)
        p[i] = 0.0f;
}

__global__ void cvt_bf16_kernel(const float* __restrict__ src, bf16t* __restrict__ dst, long long n)
{
    long long stride = (long long)gridDim.x * blockDim.x;
    for (long long i = blockIdx.x * (long long)blockDim.x + threadIdx.x; i < n; i += stride)
        dst[i] = (bf16t)src[i];
}

// dst: mats x [rowsp x cols]; src: mats x [rows x cols]; pad rows zero-filled
__global__ void cvt_pad_rows_kernel(const float* __restrict__ src, bf16t* __restrict__ dst,
                                    int mats, int rows, int rowsp, int cols)
{
    long long total  = (long long)mats * rowsp * cols;
    long long stride = (long long)gridDim.x * blockDim.x;
    for (long long idx = blockIdx.x * (long long)blockDim.x + threadIdx.x; idx < total; idx += stride) {
        int c  = (int)(idx % cols);
        long long rm = idx / cols;
        int rp = (int)(rm % rowsp);
        int m  = (int)(rm / rowsp);
        dst[idx] = (rp < rows) ? (bf16t)src[((long long)m * rows + rp) * cols + c] : (bf16t)0.0f;
    }
}

// dst: B x [rowsp x colsp] zero-padded both dims; src: B x [rows x cols]
__global__ void cvt_pad2d_kernel(const float* __restrict__ src, bf16t* __restrict__ dst,
                                 int batch, int rows, int cols, int rowsp, int colsp)
{
    long long total  = (long long)batch * rowsp * colsp;
    long long stride = (long long)gridDim.x * blockDim.x;
    for (long long idx = blockIdx.x * (long long)blockDim.x + threadIdx.x; idx < total; idx += stride) {
        int j  = (int)(idx % colsp);
        long long rm = idx / colsp;
        int i  = (int)(rm % rowsp);
        long long b = rm / rowsp;
        float v = (i < rows && j < cols) ? src[(b * rows + i) * (long long)cols + j] : 0.0f;
        dst[idx] = (bf16t)v;
    }
}

__global__ void scatter_adj_kernel(const int* __restrict__ ei, const float* __restrict__ ea,
                                   float* __restrict__ A, int E)
{
    int e = blockIdx.x * blockDim.x + threadIdx.x;
    if (e >= E) return;
    int s = ei[e];          // global src
    int d = ei[E + e];      // global dst
    int g = s / 200;
    atomicAdd(&A[(long long)g * 200 * 200 + (long long)(d % 200) * 200 + (s % 200)], ea[e]);
}

// one wave per adjacency row: dis = deg>0 ? rsqrt(deg) : 0
__global__ void deg_dis_kernel(int n, long long rows, const float* __restrict__ A,
                               float* __restrict__ dis)
{
    long long row = (long long)blockIdx.x * (blockDim.x / 32) + (threadIdx.x >> 5);
    int lane = threadIdx.x & 31;
    if (row >= rows) return;
    const float* Ar = A + row * n;
    float s = 0.0f;
    for (int j = lane; j < n; j += 32) s += Ar[j];
    for (int m = 16; m; m >>= 1) s += __shfl_xor(s, m, 32);
    if (lane == 0) dis[row] = (s > 0.0f) ? rsqrtf(s) : 0.0f;
}

// L[b,i,jp] = jp<n ? -A[b,i,jp]*dis[b,i]*dis[b,jp] : 0 ; L has colsp >= n (K-padded)
__global__ void make_L_kernel(int n, int colsp, const float* __restrict__ A,
                              const float* __restrict__ dis, bf16t* __restrict__ L,
                              long long total)
{
    long long stride = (long long)gridDim.x * blockDim.x;
    for (long long idx = blockIdx.x * (long long)blockDim.x + threadIdx.x; idx < total; idx += stride) {
        int j = (int)(idx % colsp);
        long long row = idx / colsp;          // b*n + i
        long long bn = row - (row % n);       // b*n
        float v = 0.0f;
        if (j < n) v = -A[row * n + j] * dis[row] * dis[bn + j];
        L[idx] = (bf16t)v;
    }
}

// Asbf[b,i,j] = (i<n && j<n) ? Ap[b,i,j] + (i==j) : 0 ; padded to np x np
__global__ void add_identity_pad_kernel(int n, int np, const float* __restrict__ Ap,
                                        bf16t* __restrict__ Asbf, long long total)
{
    long long stride = (long long)gridDim.x * blockDim.x;
    for (long long idx = blockIdx.x * (long long)blockDim.x + threadIdx.x; idx < total; idx += stride) {
        int j = (int)(idx % np);
        long long rm = idx / np;
        int i = (int)(rm % np);
        long long b = rm / np;
        float v = 0.0f;
        if (i < n && j < n) v = Ap[(b * n + i) * (long long)n + j] + (i == j ? 1.0f : 0.0f);
        Asbf[idx] = (bf16t)v;
    }
}

__global__ void zero_diag_kernel(int n, float* __restrict__ A, long long rows)
{
    long long t = (long long)blockIdx.x * blockDim.x + threadIdx.x;
    if (t >= rows) return;
    long long b = t / n;
    int i = (int)(t - b * n);
    A[(b * n + i) * (long long)n + i] = 0.0f;
}

// ---------------------------------------------------------------------------
// TopKPooling + gmp||gap readout per graph (one 256-thread block per graph)
// ---------------------------------------------------------------------------
template<bool WRITE_AP, bool WRITE_XPBF>
__global__ __launch_bounds__(256)
void topk_pool_kernel(int n, int k, int kpad,
                      const float* __restrict__ H,    // [B,n,64]
                      const float* __restrict__ Aadj, // [B,n,n]
                      const float* __restrict__ p,    // [64]
                      bf16t* __restrict__ xpbf,       // [B,kpad,64] (K-row-padded)
                      float* __restrict__ Ap,         // [B,k,k]
                      float* __restrict__ zout,       // [B,256]
                      int zoff)
{
    __shared__ float sc[256];
    __shared__ int   sidx[128];
    __shared__ float ssv[128];
    __shared__ float xs[100 * 64];
    __shared__ float pn;

    int b = blockIdx.x, tid = threadIdx.x;
    const float* Hb = H + (long long)b * n * 64;

    if (tid == 0) {
        float s = 0.0f;
        for (int c = 0; c < 64; ++c) s += p[c] * p[c];
        pn = fmaxf(sqrtf(s), 1e-12f);
    }
    __syncthreads();

    for (int i = tid; i < n; i += 256) {
        float dsum = 0.0f;
        for (int c = 0; c < 64; ++c) dsum += Hb[i * 64 + c] * p[c];
        sc[i] = 1.0f / (1.0f + __expf(-dsum / pn));
    }
    __syncthreads();

    // rank-based top-k (tie-break by lower index, matching lax.top_k ordering)
    for (int i = tid; i < n; i += 256) {
        float si = sc[i];
        int r = 0;
        for (int j = 0; j < n; ++j) {
            float sj = sc[j];
            r += (sj > si) || (sj == si && j < i);
        }
        if (r < k) { sidx[r] = i; ssv[r] = si; }
    }
    __syncthreads();

    for (int t = tid; t < k * 64; t += 256) {
        int r = t >> 6, c = t & 63;
        float v = Hb[(long long)sidx[r] * 64 + c] * ssv[r];
        xs[t] = v;
        if (WRITE_XPBF) xpbf[(long long)b * kpad * 64 + t] = (bf16t)v;
    }
    if (WRITE_XPBF) {   // zero K-pad rows
        for (int t = k * 64 + tid; t < kpad * 64; t += 256)
            xpbf[(long long)b * kpad * 64 + t] = (bf16t)0.0f;
    }
    __syncthreads();

    if (tid < 64) {   // gmp || gap readout
        float mx = -3.4e38f, sm = 0.0f;
        for (int r = 0; r < k; ++r) { float v = xs[r * 64 + tid]; mx = fmaxf(mx, v); sm += v; }
        zout[(long long)b * 256 + zoff + tid]      = mx;
        zout[(long long)b * 256 + zoff + 64 + tid] = sm / (float)k;
    }
    if (WRITE_AP) {
        for (int t = tid; t < k * k; t += 256) {
            int r = t / k, c = t - r * k;
            Ap[(long long)b * k * k + t] =
                Aadj[(long long)b * n * n + (long long)sidx[r] * n + sidx[c]];
        }
    }
}

// ---------------------------------------------------------------------------
// BatchNorm (training mode, biased var) over batch dim: 512 rows x 256 feats
// ---------------------------------------------------------------------------
__global__ __launch_bounds__(256)
void bn_kernel(const float* __restrict__ z1, const float* __restrict__ gamma,
               const float* __restrict__ beta, float* __restrict__ feat,
               bf16t* __restrict__ featbf)
{
    int f = blockIdx.x;   // feature 0..255
    int t = threadIdx.x;  // 0..255, handles rows t and t+256
    __shared__ float s1[256], s2[256];
    float a = z1[(long long)t * 256 + f];
    float c = z1[(long long)(t + 256) * 256 + f];
    s1[t] = a + c;
    s2[t] = a * a + c * c;
    __syncthreads();
    for (int m = 128; m; m >>= 1) {
        if (t < m) { s1[t] += s1[t + m]; s2[t] += s2[t + m]; }
        __syncthreads();
    }
    float mu  = s1[0] * (1.0f / 512.0f);
    float var = s2[0] * (1.0f / 512.0f) - mu * mu;
    float inv = rsqrtf(var + 1e-5f);
    float g = gamma[f], be = beta[f];
    float v0 = g * (a - mu) * inv + be;
    float v1 = g * (c - mu) * inv + be;
    feat[(long long)t * 256 + f]         = v0;
    feat[(long long)(t + 256) * 256 + f] = v1;
    featbf[(long long)t * 256 + f]         = (bf16t)v0;
    featbf[(long long)(t + 256) * 256 + f] = (bf16t)v1;
}

__global__ __launch_bounds__(256)
void rownorm_kernel(const float* __restrict__ src, float* __restrict__ dst, int cols)
{
    int r = blockIdx.x, t = threadIdx.x;
    __shared__ float s[256];
    float ss = 0.0f;
    for (int c = t; c < cols; c += 256) { float v = src[(long long)r * cols + c]; ss += v * v; }
    s[t] = ss;
    __syncthreads();
    for (int m = 128; m; m >>= 1) { if (t < m) s[t] += s[t + m]; __syncthreads(); }
    float inv = 1.0f / fmaxf(sqrtf(s[0]), 1e-12f);
    for (int c = t; c < cols; c += 256)
        dst[(long long)r * cols + c] = src[(long long)r * cols + c] * inv;
}

// ---------------------------------------------------------------------------
// Host orchestration
// ---------------------------------------------------------------------------
extern "C" void kernel_launch(void* const* d_in, const int* in_sizes, int n_in,
                              void* d_out, int out_size, void* d_ws, size_t ws_size,
                              hipStream_t stream)
{
    (void)n_in; (void)out_size; (void)ws_size;
    const int B = 512, Rn = 200, N2 = 100, K1 = 100, K2 = 50;
    const int KP1 = 224;   // 200 padded to mult of 32
    const int KP2 = 128;   // 100 padded to mult of 32
    const long long S1 = (long long)KP1 * KP1;       // 50176 elems per padded [224,224]
    const long long S2 = (long long)KP2 * KP2;       // 16384 elems per padded [128,128]
    const long long SX = (long long)KP2 * 64;        // 8192 elems per padded [128,64]

    const float* x   = (const float*)d_in[0];
    const int*   ei  = (const int*)d_in[1];
    const float* ea  = (const float*)d_in[2];
    const float* Wc1 = (const float*)d_in[4];
    const float* bc1 = (const float*)d_in[5];
    const float* p1  = (const float*)d_in[6];
    const float* Wc2 = (const float*)d_in[7];
    const float* bc2 = (const float*)d_in[8];
    const float* p2  = (const float*)d_in[9];
    const float* Wfc = (const float*)d_in[10];
    const float* bfc = (const float*)d_in[11];
    const float* gam = (const float*)d_in[12];
    const float* bet = (const float*)d_in[13];
    const float* Wm1 = (const float*)d_in[14];
    const float* Wm2 = (const float*)d_in[15];
    const float* bm2 = (const float*)d_in[16];
    float* out = (float*)d_out;
    const int E = in_sizes[2];

    // ---- workspace bump allocator ----
    char* base = (char*)d_ws;
    size_t off = 0;
    auto alloc = [&](size_t bytes) -> void* {
        void* ptr = base + off;
        off += (bytes + 255) & ~(size_t)255;
        return ptr;
    };
    float* Af     = (float*)alloc((size_t)B * Rn * Rn * 4);          // 82 MB (alive thru pool1)
    float* h1     = (float*)alloc((size_t)B * Rn * 64 * 4);          // 26 MB
    float* dis1   = (float*)alloc((size_t)B * Rn * 4);
    float* dis2   = (float*)alloc((size_t)B * N2 * 4);
    bf16t* Wc1b   = (bf16t*)alloc((size_t)3 * KP1 * 64 * 2);         // K-row-padded
    bf16t* Wc2b   = (bf16t*)alloc((size_t)3 * 64 * 64 * 2);
    bf16t* Wfcb   = (bf16t*)alloc((size_t)256 * 256 * 2);
    bf16t* Wm1b   = (bf16t*)alloc((size_t)256 * 64 * 2);
    bf16t* Wm2b   = (bf16t*)alloc((size_t)64 * 512 * 2);
    float* zb     = (float*)alloc((size_t)B * 256 * 4);
    bf16t* zbf    = (bf16t*)alloc((size_t)B * 256 * 2);
    float* z1     = (float*)alloc((size_t)B * 256 * 4);
    float* feat   = (float*)alloc((size_t)B * 256 * 4);
    bf16t* featbf = (bf16t*)alloc((size_t)B * 256 * 2);
    bf16t* t1bf   = (bf16t*)alloc((size_t)B * 64 * 2);
    float* outf   = (float*)alloc((size_t)B * 512 * 4);
    char*  big    = (char*)alloc((size_t)4 * B * S1 * 2);            // 206 MB shared region
    // stage-1 layout in `big`: four padded [B,224,224] bf16 matrices
    const size_t SZL = (size_t)B * S1 * 2;
    bf16t* Lbf   = (bf16t*)(big);
    bf16t* xbf   = (bf16t*)(big + SZL);
    bf16t* Tx1b  = (bf16t*)(big + 2 * SZL);
    bf16t* Tx2b  = (bf16t*)(big + 3 * SZL);
    // stage-2 layout aliased into `big` (stage-1 buffers dead after h1 is built)
    float* Apf   = (float*)(big);                                    // [B,100,100] f32
    bf16t* Asbf  = (bf16t*)(big + 20480000);                         // [B,128,128] bf16
    float* A2f   = (float*)(big + 37257216);                         // [B,100,100] f32
    bf16t* L2bf  = (bf16t*)(big + 57737216);                         // [B,100,128] bf16
    bf16t* xp1b  = (bf16t*)(big + 70844416);                         // [B,128,64]  bf16
    bf16t* Tx1b2 = (bf16t*)(big + 79233024);                         // [B,128,64]  bf16
    bf16t* Tx2b2 = (bf16t*)(big + 87621632);                         // [B,128,64]  bf16
    float* h2    = (float*)(big + 96010240);                         // [B,100,64]  f32

    const long long nA  = (long long)B * Rn * Rn;   // 20.48M
    dim3 blk256(256);

    // ---- build adjacency ----
    zero_f32_kernel<<<4096, blk256, 0, stream>>>(Af, nA);
    scatter_adj_kernel<<<(E + 255) / 256, blk256, 0, stream>>>(ei, ea, Af, E);

    // ---- L1 = -D^-1/2 A D^-1/2 (K-padded bf16) ; x,weights -> padded bf16 ----
    deg_dis_kernel<<<(B * Rn + 7) / 8, blk256, 0, stream>>>(Rn, (long long)B * Rn, Af, dis1);
    make_L_kernel<<<4096, blk256, 0, stream>>>(Rn, KP1, Af, dis1, Lbf, (long long)B * Rn * KP1);
    cvt_pad2d_kernel<<<4096, blk256, 0, stream>>>(x, xbf, B, Rn, Rn, KP1, KP1);
    cvt_pad_rows_kernel<<<64, blk256, 0, stream>>>(Wc1, Wc1b, 3, Rn, KP1, 64);
    cvt_bf16_kernel<<<16, blk256, 0, stream>>>(Wc2, Wc2b, 3LL * 64 * 64);
    cvt_bf16_kernel<<<64, blk256, 0, stream>>>(Wfc, Wfcb, 256LL * 256);
    cvt_bf16_kernel<<<16, blk256, 0, stream>>>(Wm1, Wm1b, 256LL * 64);
    cvt_bf16_kernel<<<32, blk256, 0, stream>>>(Wm2, Wm2b, 64LL * 512);

    // ---- ChebConv1 (all WMMA; K = 224 padded) ----
    dim3 gblk(128);
    dim3 gA(4, 4, B);          // N=200, M=200
    // Tx1 = L @ x
    gemm_bf16_wmma<true, false, false, false, false><<<gA, gblk, 0, stream>>>(
        Rn, Rn, KP1, Lbf, S1, KP1, xbf, S1, KP1, Tx1b, S1, KP1,
        nullptr, 0, 0, nullptr, 1.0f, 0.0f);
    // Tx2 = 2 L @ Tx1 - x
    gemm_bf16_wmma<true, false, true, false, false><<<gA, gblk, 0, stream>>>(
        Rn, Rn, KP1, Lbf, S1, KP1, Tx1b, S1, KP1, Tx2b, S1, KP1,
        xbf, S1, KP1, nullptr, 2.0f, -1.0f);
    // h1 = x@W0 + b ; += Tx1@W1 ; += Tx2@W2
    dim3 gH(1, 4, B);          // N=64, M=200
    gemm_bf16_wmma<false, false, false, true, false><<<gH, gblk, 0, stream>>>(
        Rn, 64, KP1, xbf, S1, KP1, Wc1b, 0, 64, h1, 12800, 64,
        nullptr, 0, 0, bc1, 1.0f, 0.0f);
    gemm_bf16_wmma<false, true, false, false, false><<<gH, gblk, 0, stream>>>(
        Rn, 64, KP1, Tx1b, S1, KP1, Wc1b + 1LL * KP1 * 64, 0, 64, h1, 12800, 64,
        nullptr, 0, 0, nullptr, 1.0f, 0.0f);
    gemm_bf16_wmma<false, true, false, false, false><<<gH, gblk, 0, stream>>>(
        Rn, 64, KP1, Tx2b, S1, KP1, Wc1b + 2LL * KP1 * 64, 0, 64, h1, 12800, 64,
        nullptr, 0, 0, nullptr, 1.0f, 0.0f);

    // ---- pool1 (+ readout into z[:,0:128]) ----
    topk_pool_kernel<true, true><<<B, blk256, 0, stream>>>(
        Rn, K1, KP2, h1, Af, p1, xp1b, Apf, zb, 0);

    // ---- augment_adj: A2 = (Ap+I)(Ap+I) with zeroed diagonal; L2 in padded bf16 ----
    add_identity_pad_kernel<<<2048, blk256, 0, stream>>>(N2, KP2, Apf, Asbf, (long long)B * S2);
    dim3 gA2(2, 2, B);
    gemm_bf16_wmma<false, false, false, false, false><<<gA2, gblk, 0, stream>>>(
        N2, N2, KP2, Asbf, S2, KP2, Asbf, S2, KP2, A2f, 10000, N2,
        nullptr, 0, 0, nullptr, 1.0f, 0.0f);
    zero_diag_kernel<<<(B * N2 + 255) / 256, blk256, 0, stream>>>(N2, A2f, (long long)B * N2);
    deg_dis_kernel<<<(B * N2 + 7) / 8, blk256, 0, stream>>>(N2, (long long)B * N2, A2f, dis2);
    make_L_kernel<<<2048, blk256, 0, stream>>>(N2, KP2, A2f, dis2, L2bf, (long long)B * N2 * KP2);

    // ---- ChebConv2 (K = 128 padded / 64 exact) ----
    dim3 gH2(1, 2, B);         // N=64, M=100
    gemm_bf16_wmma<true, false, false, false, false><<<gH2, gblk, 0, stream>>>(
        N2, 64, KP2, L2bf, (long long)N2 * KP2, KP2, xp1b, SX, 64, Tx1b2, SX, 64,
        nullptr, 0, 0, nullptr, 1.0f, 0.0f);
    gemm_bf16_wmma<true, false, true, false, false><<<gH2, gblk, 0, stream>>>(
        N2, 64, KP2, L2bf, (long long)N2 * KP2, KP2, Tx1b2, SX, 64, Tx2b2, SX, 64,
        xp1b, SX, 64, nullptr, 2.0f, -1.0f);
    gemm_bf16_wmma<false, false, false, true, false><<<gH2, gblk, 0, stream>>>(
        N2, 64, 64, xp1b, SX, 64, Wc2b, 0, 64, h2, 6400, 64,
        nullptr, 0, 0, bc2, 1.0f, 0.0f);
    gemm_bf16_wmma<false, true, false, false, false><<<gH2, gblk, 0, stream>>>(
        N2, 64, 64, Tx1b2, SX, 64, Wc2b + 1LL * 64 * 64, 0, 64, h2, 6400, 64,
        nullptr, 0, 0, nullptr, 1.0f, 0.0f);
    gemm_bf16_wmma<false, true, false, false, false><<<gH2, gblk, 0, stream>>>(
        N2, 64, 64, Tx2b2, SX, 64, Wc2b + 2LL * 64 * 64, 0, 64, h2, 6400, 64,
        nullptr, 0, 0, nullptr, 1.0f, 0.0f);

    // ---- pool2 (+ readout into z[:,128:256]) ----
    topk_pool_kernel<false, false><<<B, blk256, 0, stream>>>(
        N2, K2, K2, h2, nullptr, p2, nullptr, nullptr, zb, 128);

    // ---- FC + BN + contra encoder ----
    cvt_bf16_kernel<<<512, blk256, 0, stream>>>(zb, zbf, (long long)B * 256);
    dim3 gFC(4, 8, 1);         // M=512, N=256, K=256
    gemm_bf16_wmma<false, false, false, true, true><<<gFC, gblk, 0, stream>>>(
        B, 256, 256, zbf, 0, 256, Wfcb, 0, 256, z1, 0, 256,
        nullptr, 0, 0, bfc, 1.0f, 0.0f);
    bn_kernel<<<256, blk256, 0, stream>>>(z1, gam, bet, feat, featbf);
    dim3 gM1(1, 8, 1);         // M=512, N=64, K=256
    gemm_bf16_wmma<true, false, false, false, true><<<gM1, gblk, 0, stream>>>(
        B, 64, 256, featbf, 0, 256, Wm1b, 0, 64, t1bf, 0, 64,
        nullptr, 0, 0, nullptr, 1.0f, 0.0f);
    dim3 gM2(8, 8, 1);         // M=512, N=512, K=64
    gemm_bf16_wmma<false, false, false, true, false><<<gM2, gblk, 0, stream>>>(
        B, 512, 64, t1bf, 0, 64, Wm2b, 0, 512, outf, 0, 512,
        nullptr, 0, 0, bm2, 1.0f, 0.0f);

    // ---- row-normalized outputs: fn [512,256] then on [512,512] ----
    rownorm_kernel<<<B, blk256, 0, stream>>>(feat, out, 256);
    rownorm_kernel<<<B, blk256, 0, stream>>>(outf, out + (long long)B * 256, 512);
}
